// HallwayMemoryScalarObs_DTwithRMT_25039659336296
// MI455X (gfx1250) — compile-verified, matrix operand order checked by reading, not run on hardware
//
#include <hip/hip_runtime.h>
#include <math.h>
#include <stdint.h>

// ---------------- model constants ----------------
constexpr int kB    = 8;
constexpr int kT    = 340;
constexpr int kM    = 32;
constexpr int kH    = 512;
constexpr int kNH   = 8;
constexpr int kDH   = 64;          // kH / kNH
constexpr int kNB   = 4;
constexpr int kNA   = 4;
constexpr int kINFO = 3 * kT;      // 1020
constexpr int kL    = kINFO + 2 * kM;   // 1084
constexpr int kBL   = kB * kL;     // 8672 = 16 * 542
constexpr int kH4   = 4 * kH;      // 2048
constexpr int kLP   = 1088;        // padded L for 16B-aligned Vt rows (68*16)
constexpr int kRT   = (kL + 15) / 16;   // 68 row tiles per (b,h) in attention

// ---------------- WMMA / TDM types ----------------
typedef __bf16       v16bf __attribute__((ext_vector_type(16)));
typedef __bf16       v8bf  __attribute__((ext_vector_type(8)));
typedef float        v8f   __attribute__((ext_vector_type(8)));
typedef unsigned int v4u   __attribute__((ext_vector_type(4)));
typedef int          v8i   __attribute__((ext_vector_type(8)));
typedef int          v4i   __attribute__((ext_vector_type(4)));

union Frag16 { v16bf v; v8bf h8[2]; };

static __device__ inline v8f wmma_bf16(v16bf a, v16bf b, v8f c) {
  // D = A(16x32 bf16) * B(32x16 bf16) + C(16x16 f32)
  return __builtin_amdgcn_wmma_f32_16x16x32_bf16(
      /*neg_a=*/false, a, /*neg_b=*/false, b,
      /*c_mod=*/(short)0, c, /*reuse_a=*/false, /*reuse_b=*/false);
}

// Build a D# for a 2D bf16 tile and issue TENSOR_LOAD_TO_LDS (6-arg clang-23 form).
// Bit layout per CDNA5 ISA 8.3/8.4: group0 = {count, lds_addr, global_addr, type=2},
// group1 = {data_size, tensor_dim0/1, tile_dim0/1, tensor_dim0_stride}.
static __device__ inline void tdm_load_tile_2d(uint32_t lds_off, const void* gaddr,
                                               uint32_t tile_w, uint32_t tile_h,
                                               uint32_t tensor_w, uint32_t tensor_h,
                                               uint32_t row_stride) {
  const uint64_t ga = (uint64_t)(uintptr_t)gaddr;
  v4u g0;
  g0[0] = 1u;                                            // count=1 (valid user D#)
  g0[1] = lds_off;                                       // LDS byte address
  g0[2] = (uint32_t)(ga & 0xFFFFFFFFu);                  // global_addr[31:0]
  g0[3] = (uint32_t)((ga >> 32) & 0x01FFFFFFu) | (2u << 30);  // addr[56:32] | type=2
  v8i g1;
  g1[0] = (int)(1u << 16);                               // data_size=1 (2 bytes)
  g1[1] = (int)((tensor_w & 0xFFFFu) << 16);             // tensor_dim0[15:0]
  g1[2] = (int)(((tensor_w >> 16) & 0xFFFFu) | ((tensor_h & 0xFFFFu) << 16));
  g1[3] = (int)(((tensor_h >> 16) & 0xFFFFu) | ((tile_w & 0xFFFFu) << 16));
  g1[4] = (int)(tile_h & 0xFFFFu);                       // tile_dim1 (tile_dim2=0)
  g1[5] = (int)row_stride;                               // tensor_dim0_stride[31:0]
  g1[6] = 0;
  g1[7] = 0;
  const v4i gz4 = {0, 0, 0, 0};                          // 2D tensor: groups 2/3 unused
  const v8i gz8 = {0, 0, 0, 0, 0, 0, 0, 0};
  __builtin_amdgcn_tensor_load_to_lds(g0, g1, gz4, gz4, gz8, 0);
}

static __device__ inline bool attn_ok(int i, int j) {
  if (j < kM) return true;                 // every row attends the read-memory
  if (i >= kL - kM) return true;           // write-memory rows attend everything
  return (i >= kM) && (i < kM + kINFO) && (j >= kM) && (j <= i);  // causal core
}

static __device__ inline float gelu_exact(float x) {
  return 0.5f * x * (1.0f + erff(x * 0.70710678118654752f));
}

// ---------------- tiny helpers ----------------
__global__ void cvt_f32_to_bf16(const float* __restrict__ s, __bf16* __restrict__ d, int n) {
  for (int i = blockIdx.x * blockDim.x + threadIdx.x; i < n; i += gridDim.x * blockDim.x)
    d[i] = (__bf16)s[i];
}

__global__ void zero_bf16_kernel(__bf16* __restrict__ d, int n) {
  for (int i = blockIdx.x * blockDim.x + threadIdx.x; i < n; i += gridDim.x * blockDim.x)
    d[i] = (__bf16)0.0f;
}

// ---------------- embedding (raw hidden, pre-LN) ----------------
__global__ __launch_bounds__(128)
void embed_kernel(const float* __restrict__ returns_, const float* __restrict__ actions_,
                  const float* __restrict__ t_table, const float* __restrict__ s_table,
                  const float* __restrict__ Wr, const float* __restrict__ br,
                  const float* __restrict__ Wa, const float* __restrict__ ba,
                  const float* __restrict__ read_mem, const float* __restrict__ mem_tok,
                  const int* __restrict__ states, const int* __restrict__ timestep,
                  float* __restrict__ out) {
  const int l = blockIdx.x, b = blockIdx.y, tid = threadIdx.x;
  float* o = out + ((size_t)b * kL + l) * kH;
  if (l < kM) {
    #pragma unroll
    for (int i = 0; i < 4; i++) { int c = tid + i * 128; o[c] = read_mem[l * kH + c]; }
  } else if (l >= kM + kINFO) {
    const int mi = l - kM - kINFO;
    #pragma unroll
    for (int i = 0; i < 4; i++) { int c = tid + i * 128; o[c] = mem_tok[mi * kH + c]; }
  } else {
    const int kk = l - kM, t = kk / 3, c3 = kk % 3;
    if (c3 == 1) {
      const int st = states[b * kT + t];
      #pragma unroll
      for (int i = 0; i < 4; i++) { int c = tid + i * 128; o[c] = s_table[(size_t)st * kH + c]; }
    } else {
      const int ts = timestep[b * kT + t];
      const float val = (c3 == 0) ? returns_[b * kT + t] : actions_[b * kT + t];
      const float* Wx = (c3 == 0) ? Wr : Wa;
      const float* bx = (c3 == 0) ? br : ba;
      #pragma unroll
      for (int i = 0; i < 4; i++) {
        int c = tid + i * 128;
        o[c] = val * Wx[c] + bx[c] + t_table[(size_t)ts * kH + c];
      }
    }
  }
}

// ---------------- LayerNorm over H=512, one row per 128-thread block ----------------
__global__ __launch_bounds__(128)
void ln_kernel(const float* __restrict__ src, const float* __restrict__ g,
               const float* __restrict__ be, float* __restrict__ outf,
               __bf16* __restrict__ outb) {
  __shared__ float red[4];
  const int row = blockIdx.x, tid = threadIdx.x;
  const float* x = src + (size_t)row * kH;
  float v[4], s = 0.0f;
  #pragma unroll
  for (int i = 0; i < 4; i++) { v[i] = x[tid + i * 128]; s += v[i]; }
  #pragma unroll
  for (int d = 1; d < 32; d <<= 1) s += __shfl_xor(s, d, 32);
  if ((tid & 31) == 0) red[tid >> 5] = s;
  __syncthreads();
  const float mu = (red[0] + red[1] + red[2] + red[3]) * (1.0f / (float)kH);
  float vs = 0.0f;
  #pragma unroll
  for (int i = 0; i < 4; i++) { float d0 = v[i] - mu; vs += d0 * d0; }
  #pragma unroll
  for (int d = 1; d < 32; d <<= 1) vs += __shfl_xor(vs, d, 32);
  __syncthreads();
  if ((tid & 31) == 0) red[tid >> 5] = vs;
  __syncthreads();
  const float rstd = rsqrtf((red[0] + red[1] + red[2] + red[3]) * (1.0f / (float)kH) + 1e-5f);
  #pragma unroll
  for (int i = 0; i < 4; i++) {
    int c = tid + i * 128;
    float y = (v[i] - mu) * rstd * g[c] + be[c];
    outf[(size_t)row * kH + c] = y;
    outb[(size_t)row * kH + c] = (__bf16)y;
  }
}

// ---------------- WMMA GEMM: Y = X(BLxK) @ W^T(NxK) + bias -----------------
// MODE 0: store bf16            MODE 1: GELU(exact) -> bf16
// MODE 2: +resid  -> f32        MODE 3: store bf16 transposed into Vt(B,NH,DH,LP)
template <int MODE>
__global__ __launch_bounds__(256)
void gemm_bf16_kernel(const __bf16* __restrict__ X, const __bf16* __restrict__ W,
                      const float* __restrict__ bias, const float* __restrict__ resid,
                      float* __restrict__ out_f32, __bf16* __restrict__ out_bf,
                      int K, int N) {
  __shared__ __align__(16) __bf16 As[16][32];
  const int tid  = threadIdx.x;
  const int lane = tid & 31;
  const int w    = tid >> 5;                     // 8 waves
  const int rowbase = blockIdx.x * 16;
  const int colbase = blockIdx.y * 512 + w * 64; // 4 x 16-col tiles per wave
  const int arow  = lane & 15;
  const int khalf = lane >> 4;
  const uint32_t lds_off = (uint32_t)(uintptr_t)&As[0][0];

  v8f acc0 = {}, acc1 = {}, acc2 = {}, acc3 = {};
  for (int k0 = 0; k0 < K; k0 += 32) {
    // Tensor Data Mover: DMA the 16x32 bf16 A tile from global into LDS.
    // TDM ignores EXEC and issues once per wave -> restrict to one wave.
    if (tid == 0) {
      tdm_load_tile_2d(lds_off, X + (size_t)rowbase * K + k0,
                       /*tile_w=*/32, /*tile_h=*/16,
                       /*tensor_w=*/(uint32_t)(K - k0),
                       /*tensor_h=*/(uint32_t)(kBL - rowbase),
                       /*row_stride=*/(uint32_t)K);
      __builtin_amdgcn_s_wait_tensorcnt(0);
    }
    __syncthreads();

    Frag16 a;
    a.h8[0] = *(const v8bf*)&As[arow][khalf * 8];
    a.h8[1] = *(const v8bf*)&As[arow][16 + khalf * 8];

    const int kpf = (k0 + 32 < K) ? (k0 + 32) : k0;  // branchless prefetch addr
    #pragma unroll
    for (int t = 0; t < 4; t++) {
      const __bf16* wr = W + (size_t)(colbase + t * 16 + arow) * K + k0;
      Frag16 bfr;
      bfr.h8[0] = *(const v8bf*)(wr + khalf * 8);
      bfr.h8[1] = *(const v8bf*)(wr + 16 + khalf * 8);
      __builtin_prefetch(wr - k0 + kpf, 0, 3);       // next K-slab of this W row
      v8f& acc = (t == 0) ? acc0 : (t == 1) ? acc1 : (t == 2) ? acc2 : acc3;
      acc = wmma_bf16(a.v, bfr.v, acc);
    }
    __syncthreads();
  }

  const int rowoff = khalf * 8;
  v8f accs[4] = {acc0, acc1, acc2, acc3};
  #pragma unroll
  for (int t = 0; t < 4; t++) {
    const int col = colbase + t * 16 + arow;
    const float bv = bias[col];
    #pragma unroll
    for (int r = 0; r < 8; r++) {
      const int row = rowbase + r + rowoff;
      float y = accs[t][r] + bv;
      if (MODE == 1) y = gelu_exact(y);
      if (MODE == 2) {
        y += resid[(size_t)row * N + col];
        out_f32[(size_t)row * N + col] = y;
      } else if (MODE == 3) {
        const int bb = row / kL, token = row % kL;
        const int hh = col >> 6, d = col & 63;
        out_bf[(((size_t)bb * kNH + hh) * kDH + d) * kLP + token] = (__bf16)y;
      } else {
        out_bf[(size_t)row * N + col] = (__bf16)y;
      }
    }
  }
}

// ---------------- flash attention: one wave per (b, h, 16-row tile) ----------------
__global__ __launch_bounds__(32)
void attn_kernel(const __bf16* __restrict__ q, const __bf16* __restrict__ k,
                 const __bf16* __restrict__ vt, __bf16* __restrict__ out) {
  __shared__ __align__(16) __bf16 P[16][32];
  const int lane  = threadIdx.x;
  const int arow  = lane & 15;
  const int khalf = lane >> 4;
  const int rt = blockIdx.x, h = blockIdx.y, b = blockIdx.z;
  const int rowbase = rt * 16;

  // Q fragments for head-dim K = 0..31 and 32..63
  int qi = rowbase + arow; if (qi >= kL) qi = kL - 1;
  const __bf16* qp = q + ((size_t)(b * kL + qi)) * kH + h * kDH;
  Frag16 qa0, qa1;
  qa0.h8[0] = *(const v8bf*)(qp + khalf * 8);
  qa0.h8[1] = *(const v8bf*)(qp + 16 + khalf * 8);
  qa1.h8[0] = *(const v8bf*)(qp + 32 + khalf * 8);
  qa1.h8[1] = *(const v8bf*)(qp + 48 + khalf * 8);

  v8f accv[4] = {};
  float m_run[8], l_run[8];
  #pragma unroll
  for (int r = 0; r < 8; r++) { m_run[r] = -INFINITY; l_run[r] = 0.0f; }
  const float scale = 0.125f;  // 1/sqrt(64)

  for (int jc = 0; jc < kLP / 32; jc++) {
    const int j0 = jc * 32;
    Frag16 kb[2][2];
    #pragma unroll
    for (int tj = 0; tj < 2; tj++) {
      int kj = j0 + tj * 16 + arow; if (kj >= kL) kj = kL - 1;
      const __bf16* kp = k + ((size_t)(b * kL + kj)) * kH + h * kDH;
      kb[tj][0].h8[0] = *(const v8bf*)(kp + khalf * 8);
      kb[tj][0].h8[1] = *(const v8bf*)(kp + 16 + khalf * 8);
      kb[tj][1].h8[0] = *(const v8bf*)(kp + 32 + khalf * 8);
      kb[tj][1].h8[1] = *(const v8bf*)(kp + 48 + khalf * 8);
    }
    v8f s0 = {}, s1 = {};
    s0 = wmma_bf16(qa0.v, kb[0][0].v, s0);
    s0 = wmma_bf16(qa1.v, kb[0][1].v, s0);
    s1 = wmma_bf16(qa0.v, kb[1][0].v, s1);
    s1 = wmma_bf16(qa1.v, kb[1][1].v, s1);

    float p0[8], p1[8];
    #pragma unroll
    for (int r = 0; r < 8; r++) {
      const int row = rowbase + r + khalf * 8;
      const int ja = j0 + arow, jb = j0 + 16 + arow;
      const float sa = s0[r] * scale, sb = s1[r] * scale;
      const bool oka = (ja < kL) && attn_ok(row, ja);
      const bool okb = (jb < kL) && attn_ok(row, jb);
      float mc = fmaxf(oka ? sa : -INFINITY, okb ? sb : -INFINITY);
      #pragma unroll
      for (int d = 1; d < 16; d <<= 1) mc = fmaxf(mc, __shfl_xor(mc, d, 16));
      const float mn = fmaxf(m_run[r], mc);
      const float f  = (m_run[r] >= mn) ? 1.0f : __expf(m_run[r] - mn);
      const float pa = oka ? __expf(sa - mn) : 0.0f;
      const float pb = okb ? __expf(sb - mn) : 0.0f;
      float rs = pa + pb;
      #pragma unroll
      for (int d = 1; d < 16; d <<= 1) rs += __shfl_xor(rs, d, 16);
      l_run[r] = l_run[r] * f + rs;
      m_run[r] = mn;
      #pragma unroll
      for (int t = 0; t < 4; t++) accv[t][r] *= f;
      p0[r] = pa; p1[r] = pb;
    }

    // C-layout -> A-layout conversion of P through LDS
    #pragma unroll
    for (int r = 0; r < 8; r++) {
      const int rl = r + khalf * 8;
      P[rl][arow]      = (__bf16)p0[r];
      P[rl][16 + arow] = (__bf16)p1[r];
    }
    __syncthreads();
    Frag16 pa_;
    pa_.h8[0] = *(const v8bf*)&P[arow][khalf * 8];
    pa_.h8[1] = *(const v8bf*)&P[arow][16 + khalf * 8];
    __syncthreads();

    // O += P(16x32) * V(32x64): Vt rows are head dims -> contiguous 16B loads
    #pragma unroll
    for (int t = 0; t < 4; t++) {
      const __bf16* vp =
          vt + (((size_t)(b * kNH + h)) * kDH + t * 16 + arow) * kLP + j0;
      Frag16 vb_;
      vb_.h8[0] = *(const v8bf*)(vp + khalf * 8);
      vb_.h8[1] = *(const v8bf*)(vp + 16 + khalf * 8);
      accv[t] = wmma_bf16(pa_.v, vb_.v, accv[t]);
    }
  }

  #pragma unroll
  for (int r = 0; r < 8; r++) {
    const int row = rowbase + r + khalf * 8;
    if (row >= kL) continue;
    const float inv = (l_run[r] > 0.0f) ? 1.0f / l_run[r] : 0.0f;
    #pragma unroll
    for (int t = 0; t < 4; t++)
      out[((size_t)(b * kL + row)) * kH + h * kDH + t * 16 + arow] =
          (__bf16)(accv[t][r] * inv);
  }
}

// ---------------- output heads ----------------
__global__ __launch_bounds__(128)
void logits_kernel(const float* __restrict__ hid, const float* __restrict__ Wp,
                   const float* __restrict__ bp, float* __restrict__ out) {
  __shared__ float red[4][4];
  const int bt = blockIdx.x, b = bt / kT, t = bt % kT, tid = threadIdx.x;
  const float* x = hid + ((size_t)(b * kL) + kM + 3 * t + 1) * kH;
  float acc[4] = {0.f, 0.f, 0.f, 0.f};
  #pragma unroll
  for (int i = 0; i < 4; i++) {
    const int c = tid + i * 128;
    const float xv = x[c];
    #pragma unroll
    for (int a = 0; a < 4; a++) acc[a] += xv * Wp[a * kH + c];
  }
  #pragma unroll
  for (int a = 0; a < 4; a++) {
    float s = acc[a];
    #pragma unroll
    for (int d = 1; d < 32; d <<= 1) s += __shfl_xor(s, d, 32);
    if ((tid & 31) == 0) red[tid >> 5][a] = s;
  }
  __syncthreads();
  if (tid < 4)
    out[(size_t)bt * kNA + tid] =
        red[0][tid] + red[1][tid] + red[2][tid] + red[3][tid] + bp[tid];
}

__global__ void memout_kernel(const float* __restrict__ hid, float* __restrict__ out) {
  const int n = kB * kM * kH;
  for (int i = blockIdx.x * blockDim.x + threadIdx.x; i < n; i += gridDim.x * blockDim.x) {
    const int b = i / (kM * kH), rem = i % (kM * kH), m = rem / kH, c = rem % kH;
    out[i] = hid[((size_t)b * kL + (kL - kM) + m) * kH + c];
  }
}

// ---------------- host orchestration ----------------
extern "C" void kernel_launch(void* const* d_in, const int* in_sizes, int n_in,
                              void* d_out, int out_size, void* d_ws, size_t ws_size,
                              hipStream_t stream) {
  (void)in_sizes; (void)n_in; (void)out_size; (void)ws_size;

  const float* returns_ = (const float*)d_in[0];
  const float* actions_ = (const float*)d_in[1];
  const float* t_table  = (const float*)d_in[2];
  const float* s_table  = (const float*)d_in[3];
  const float* Wr   = (const float*)d_in[4];
  const float* br   = (const float*)d_in[5];
  const float* Wa   = (const float*)d_in[6];
  const float* ba   = (const float*)d_in[7];
  const float* ln_e_g = (const float*)d_in[8];
  const float* ln_e_b = (const float*)d_in[9];
  const float* read_mem = (const float*)d_in[10];
  const float* mem_tok  = (const float*)d_in[11];
  const float* Wq = (const float*)d_in[12];
  const float* bq = (const float*)d_in[13];
  const float* Wk = (const float*)d_in[14];
  const float* bk = (const float*)d_in[15];
  const float* Wv = (const float*)d_in[16];
  const float* bv = (const float*)d_in[17];
  const float* Wo = (const float*)d_in[18];
  const float* bo = (const float*)d_in[19];
  const float* W1 = (const float*)d_in[20];
  const float* b1 = (const float*)d_in[21];
  const float* W2 = (const float*)d_in[22];
  const float* b2 = (const float*)d_in[23];
  const float* g1  = (const float*)d_in[24];
  const float* be1 = (const float*)d_in[25];
  const float* g2  = (const float*)d_in[26];
  const float* be2 = (const float*)d_in[27];
  const float* Wp = (const float*)d_in[28];
  const float* bp = (const float*)d_in[29];
  const int* states   = (const int*)d_in[30];
  const int* timestep = (const int*)d_in[31];

  // workspace carve-up
  size_t off = 0;
  auto alloc = [&](size_t bytes) -> void* {
    void* p = (char*)d_ws + off;
    off = (off + bytes + 255) & ~(size_t)255;
    return p;
  };
  float*  tmp_f32 = (float*)alloc((size_t)kBL * kH * 4);
  float*  h_f32   = (float*)alloc((size_t)kBL * kH * 4);
  float*  ln1_f32 = (float*)alloc((size_t)kBL * kH * 4);
  __bf16* x_bf    = (__bf16*)alloc((size_t)kBL * kH * 2);
  __bf16* q_bf    = (__bf16*)alloc((size_t)kBL * kH * 2);
  __bf16* k_bf    = (__bf16*)alloc((size_t)kBL * kH * 2);
  __bf16* attn_bf = (__bf16*)alloc((size_t)kBL * kH * 2);
  __bf16* ln1_bf  = (__bf16*)alloc((size_t)kBL * kH * 2);
  __bf16* vt_bf   = (__bf16*)alloc((size_t)kB * kNH * kDH * kLP * 2);
  __bf16* mid_bf  = (__bf16*)alloc((size_t)kBL * kH4 * 2);
  __bf16* wq_b = (__bf16*)alloc((size_t)kNB * kH * kH * 2);
  __bf16* wk_b = (__bf16*)alloc((size_t)kNB * kH * kH * 2);
  __bf16* wv_b = (__bf16*)alloc((size_t)kNB * kH * kH * 2);
  __bf16* wo_b = (__bf16*)alloc((size_t)kNB * kH * kH * 2);
  __bf16* w1_b = (__bf16*)alloc((size_t)kNB * kH4 * kH * 2);
  __bf16* w2_b = (__bf16*)alloc((size_t)kNB * kH * kH4 * 2);

  // weight conversion to bf16 (deterministic each call)
  const int nHH = kNB * kH * kH, nH4 = kNB * kH4 * kH;
  cvt_f32_to_bf16<<<1024, 256, 0, stream>>>(Wq, wq_b, nHH);
  cvt_f32_to_bf16<<<1024, 256, 0, stream>>>(Wk, wk_b, nHH);
  cvt_f32_to_bf16<<<1024, 256, 0, stream>>>(Wv, wv_b, nHH);
  cvt_f32_to_bf16<<<1024, 256, 0, stream>>>(Wo, wo_b, nHH);
  cvt_f32_to_bf16<<<1024, 256, 0, stream>>>(W1, w1_b, nH4);
  cvt_f32_to_bf16<<<1024, 256, 0, stream>>>(W2, w2_b, nH4);
  zero_bf16_kernel<<<1024, 256, 0, stream>>>(vt_bf, kB * kNH * kDH * kLP);

  // embedding + input LayerNorm
  embed_kernel<<<dim3(kL, kB), 128, 0, stream>>>(
      returns_, actions_, t_table, s_table, Wr, br, Wa, ba,
      read_mem, mem_tok, states, timestep, tmp_f32);
  ln_kernel<<<kBL, 128, 0, stream>>>(tmp_f32, ln_e_g, ln_e_b, h_f32, x_bf);

  const dim3 gemmBlk(256);
  const dim3 grid512(kBL / 16, kH / 512);    // N = 512
  const dim3 grid2048(kBL / 16, kH4 / 512);  // N = 2048
  const dim3 attnGrid(kRT, kNH, kB);

  for (int nb = 0; nb < kNB; nb++) {
    const __bf16* wqn = wq_b + (size_t)nb * kH * kH;
    const __bf16* wkn = wk_b + (size_t)nb * kH * kH;
    const __bf16* wvn = wv_b + (size_t)nb * kH * kH;
    const __bf16* won = wo_b + (size_t)nb * kH * kH;
    const __bf16* w1n = w1_b + (size_t)nb * kH4 * kH;
    const __bf16* w2n = w2_b + (size_t)nb * kH * kH4;

    gemm_bf16_kernel<0><<<grid512, gemmBlk, 0, stream>>>(
        x_bf, wqn, bq + nb * kH, nullptr, nullptr, q_bf, kH, kH);
    gemm_bf16_kernel<0><<<grid512, gemmBlk, 0, stream>>>(
        x_bf, wkn, bk + nb * kH, nullptr, nullptr, k_bf, kH, kH);
    gemm_bf16_kernel<3><<<grid512, gemmBlk, 0, stream>>>(
        x_bf, wvn, bv + nb * kH, nullptr, nullptr, vt_bf, kH, kH);

    attn_kernel<<<attnGrid, 32, 0, stream>>>(q_bf, k_bf, vt_bf, attn_bf);

    gemm_bf16_kernel<2><<<grid512, gemmBlk, 0, stream>>>(
        attn_bf, won, bo + nb * kH, h_f32, tmp_f32, nullptr, kH, kH);
    ln_kernel<<<kBL, 128, 0, stream>>>(tmp_f32, g1 + nb * kH, be1 + nb * kH,
                                       ln1_f32, ln1_bf);

    gemm_bf16_kernel<1><<<grid2048, gemmBlk, 0, stream>>>(
        ln1_bf, w1n, b1 + nb * kH4, nullptr, nullptr, mid_bf, kH, kH4);
    gemm_bf16_kernel<2><<<grid512, gemmBlk, 0, stream>>>(
        mid_bf, w2n, b2 + nb * kH, ln1_f32, tmp_f32, nullptr, kH4, kH);
    ln_kernel<<<kBL, 128, 0, stream>>>(tmp_f32, g2 + nb * kH, be2 + nb * kH,
                                       h_f32, x_bf);
  }

  float* out_logits = (float*)d_out;
  float* out_mem    = out_logits + (size_t)kB * kT * kNA;
  logits_kernel<<<kB * kT, 128, 0, stream>>>(h_f32, Wp, bp, out_logits);
  memout_kernel<<<512, 256, 0, stream>>>(h_f32, out_mem);
}
